// BinaryDense_39573828666262
// MI455X (gfx1250) — 8-bit matrix operand order repaired and checked by reading, compiled, not run on hardware
//
#include <hip/hip_runtime.h>
#include <cstdint>


#define NROWS 8192
#define KDIM  4096
#define UNITS 4096

#define TILE_M 128
#define TILE_N 128
#define TILE_K 128

typedef int   v16i __attribute__((ext_vector_type(16)));
typedef float v8f  __attribute__((ext_vector_type(8)));

__device__ __forceinline__ uint8_t sgn_fp8(float f) {
    return (f > 0.0f) ? (uint8_t)0x38 : ((f < 0.0f) ? (uint8_t)0xB8 : (uint8_t)0x00);
}

__global__ __launch_bounds__(256) void bin_x_kernel(const float* __restrict__ x,
                                                    uint8_t* __restrict__ a8) {
    const int i = blockIdx.x * blockDim.x + threadIdx.x;
    const float4 v = ((const float4*)x)[i];
    uint32_t p = ((uint32_t)sgn_fp8(v.x)) |
                 ((uint32_t)sgn_fp8(v.y) << 8) |
                 ((uint32_t)sgn_fp8(v.z) << 16) |
                 ((uint32_t)sgn_fp8(v.w) << 24);
    ((uint32_t*)a8)[i] = p;
}

__global__ __launch_bounds__(256) void bin_wT_kernel(const float* __restrict__ w,
                                                     uint8_t* __restrict__ wT) {
    __shared__ __align__(16) uint8_t tile[64 * 64];
    const int n0 = blockIdx.x * 64;
    const int k0 = blockIdx.y * 64;
    const int t  = threadIdx.x;

    const int c = (t & 15) * 4;
    const int r = t >> 4;
#pragma unroll
    for (int i = 0; i < 4; ++i) {
        const int kk = r + i * 16;
        const float4 v = *(const float4*)&w[(size_t)(k0 + kk) * UNITS + n0 + c];
        tile[(c + 0) * 64 + kk] = sgn_fp8(v.x);
        tile[(c + 1) * 64 + kk] = sgn_fp8(v.y);
        tile[(c + 2) * 64 + kk] = sgn_fp8(v.z);
        tile[(c + 3) * 64 + kk] = sgn_fp8(v.w);
    }
    __syncthreads();
#pragma unroll
    for (int i = 0; i < 4; ++i) {
        const int idx = t + 256 * i;
        const int nr = idx >> 4;
        const int kd = (idx & 15) * 4;
        *(uint32_t*)&wT[(size_t)(n0 + nr) * KDIM + k0 + kd] =
            *(const uint32_t*)&tile[nr * 64 + kd];
    }
}

__global__ __launch_bounds__(256) void bgemm_kernel(const uint8_t* __restrict__ A8,
                                                    const uint8_t* __restrict__ B8T,
                                                    float* __restrict__ out) {
    __shared__ __align__(16) uint8_t As[2][TILE_M * TILE_K];
    __shared__ __align__(16) uint8_t Bs[2][TILE_N * TILE_K];

    const int tid  = threadIdx.x;
    const int lane = tid & 31;
    const int wave = tid >> 5;
    const int wm   = wave & 3;
    const int wn   = wave >> 2;

    const int tileM = blockIdx.y * TILE_M;
    const int tileN = blockIdx.x * TILE_N;

    const int mrow = lane & 15;
    const int ksel = lane >> 4;

    v8f acc[2][4];
#pragma unroll
    for (int mi = 0; mi < 2; ++mi)
#pragma unroll
        for (int ni = 0; ni < 4; ++ni)
            acc[mi][ni] = (v8f)(0.0f);

    const int lrow = tid >> 1;
    const int lkb  = (tid & 1) * 64;
    const uint8_t* gA = &A8[(size_t)(tileM + lrow) * KDIM + lkb];
    const uint8_t* gB = &B8T[(size_t)(tileN + lrow) * KDIM + lkb];

#pragma unroll
    for (int j = 0; j < 4; ++j) {
        *(int4*)&As[0][lrow * TILE_K + lkb + j * 16] = *(const int4*)(gA + j * 16);
        *(int4*)&Bs[0][lrow * TILE_K + lkb + j * 16] = *(const int4*)(gB + j * 16);
    }
    __syncthreads();

    const int NK = KDIM / TILE_K;
    for (int it = 0; it < NK; ++it) {
        const int cur = it & 1;
        const int nxt = cur ^ 1;
        const bool has_next = (it + 1) < NK;

        int4 ra[4], rb[4];
        if (has_next) {
            const size_t koff = (size_t)(it + 1) * TILE_K;
#pragma unroll
            for (int j = 0; j < 4; ++j) {
                ra[j] = *(const int4*)(gA + koff + j * 16);
                rb[j] = *(const int4*)(gB + koff + j * 16);
            }
        }

        v16i afrag[2];
#pragma unroll
        for (int mi = 0; mi < 2; ++mi) {
            const uint8_t* p =
                &As[cur][(wm * 32 + mi * 16 + mrow) * TILE_K + ksel * 16];
            v16i a;
#pragma unroll
            for (int h = 0; h < 2; ++h) {
#pragma unroll
                for (int g = 0; g < 4; ++g) {
                    const uint2 q = *(const uint2*)(p + h * 64 + (g >> 1) * 32 + (g & 1) * 8);
                    a[h * 8 + g * 2 + 0] = (int)q.x;
                    a[h * 8 + g * 2 + 1] = (int)q.y;
                }
            }
            afrag[mi] = a;
        }

        v16i bfrag[4];
#pragma unroll
        for (int ni = 0; ni < 4; ++ni) {
            const uint8_t* p =
                &Bs[cur][(wn * 64 + ni * 16 + mrow) * TILE_K + ksel * 16];
            v16i b;
#pragma unroll
            for (int h = 0; h < 4; ++h) {
                const int4 q = *(const int4*)(p + h * 32);
                b[h * 4 + 0] = q.x; b[h * 4 + 1] = q.y;
                b[h * 4 + 2] = q.z; b[h * 4 + 3] = q.w;
            }
            bfrag[ni] = b;
        }

#pragma unroll
        for (int mi = 0; mi < 2; ++mi)
#pragma unroll
            for (int ni = 0; ni < 4; ++ni)
                acc[mi][ni] = __builtin_amdgcn_wmma_f32_16x16x128_fp8_fp8(
                    afrag[mi], bfrag[ni], (short)0, acc[mi][ni],
                     false,  false);

        if (has_next) {
#pragma unroll
            for (int j = 0; j < 4; ++j) {
                *(int4*)&As[nxt][lrow * TILE_K + lkb + j * 16] = ra[j];
                *(int4*)&Bs[nxt][lrow * TILE_K + lkb + j * 16] = rb[j];
            }
        }
        __syncthreads();
    }

#pragma unroll
    for (int mi = 0; mi < 2; ++mi) {
#pragma unroll
        for (int ni = 0; ni < 4; ++ni) {
#pragma unroll
            for (int r = 0; r < 8; ++r) {
                const int row = tileM + wm * 32 + mi * 16 + r + ksel * 8;
                const int col = tileN + wn * 64 + ni * 16 + mrow;
                out[(size_t)row * UNITS + col] = acc[mi][ni][r];
            }
        }
    }
}

extern "C" void kernel_launch(void* const* d_in, const int* in_sizes, int n_in,
                              void* d_out, int out_size, void* d_ws, size_t ws_size,
                              hipStream_t stream) {
    const float* x = (const float*)d_in[0];
    const float* w = (const float*)d_in[1];
    float* out = (float*)d_out;

    uint8_t* a8 = (uint8_t*)d_ws;
    uint8_t* wT = a8 + (size_t)NROWS * KDIM;

    bin_x_kernel<<<(NROWS * (size_t)KDIM / 4 + 255) / 256, 256, 0, stream>>>(x, a8);

    bin_wT_kernel<<<dim3(UNITS / 64, KDIM / 64), 256, 0, stream>>>(w, wT);

    bgemm_kernel<<<dim3(UNITS / TILE_N, NROWS / TILE_M), 256, 0, stream>>>(a8, wT, out);
}
